// GGLR_52630529245706
// MI455X (gfx1250) — compile-verified, matrix operand order checked by reading, not run on hardware
//
#include <hip/hip_runtime.h>
#include <hip/hip_bf16.h>
#include <stdint.h>
#include <stddef.h>

// ---------------------------------------------------------------------------
// GGLR forward for MI455X (gfx1250, wave32, WMMA).
// Heavy GEMMs run as bf16 WMMA (v_wmma_f32_16x16x32_bf16) with f32 accum.
// Staging uses native bf16 converts + global_prefetch of the A stream.
// ---------------------------------------------------------------------------

typedef __bf16 bf16_t;
typedef __attribute__((ext_vector_type(16))) __bf16 v16bf;
typedef __attribute__((ext_vector_type(8)))  __bf16 v8bf;
typedef __attribute__((ext_vector_type(4)))  __bf16 v4bf;
typedef __attribute__((ext_vector_type(8)))  float  v8f;
typedef __attribute__((ext_vector_type(4)))  float  v4f;

#define NTOT 8192
#define DM   128
#define KLAY 3
#define LDS_STRIDE 40   // 32 K-halves + 8 pad (keeps 16B alignment: 40*2=80B)

__device__ __forceinline__ bf16_t f2bf(float f) {
#if defined(__AMDGCN__)
    return (bf16_t)f;                              // native v_cvt (RNE) on gfx1250
#else
    uint32_t u = __builtin_bit_cast(uint32_t, f);
    uint32_t r = u + 0x7FFFu + ((u >> 16) & 1u);
    uint16_t h = (uint16_t)(r >> 16);
    return __builtin_bit_cast(bf16_t, h);
#endif
}

// -------------------------- degree reductions ------------------------------
__global__ __launch_bounds__(256) void colsum_kernel(const float* __restrict__ A,
                                                     float* __restrict__ Dout) {
    int j = blockIdx.x * blockDim.x + threadIdx.x;   // coalesced down columns
    if (j >= NTOT) return;
    float s = 0.f;
    for (int i = 0; i < NTOT; ++i) s += A[(size_t)i * NTOT + j];
    Dout[j] = s;
}

__global__ __launch_bounds__(256) void rowsum_kernel(const float* __restrict__ A,
                                                     float* __restrict__ Din) {
    __shared__ float red[256];
    int row = blockIdx.x;
    float s = 0.f;
    for (int j = threadIdx.x; j < NTOT; j += 256) s += A[(size_t)row * NTOT + j];
    red[threadIdx.x] = s;
    __syncthreads();
    for (int w = 128; w > 0; w >>= 1) {
        if (threadIdx.x < (unsigned)w) red[threadIdx.x] += red[threadIdx.x + w];
        __syncthreads();
    }
    if (threadIdx.x == 0) Din[row] = red[0];
}

// --------------- per-layer Linear on ORIGINAL p/q -> bf16 [N][3*128] -------
__global__ __launch_bounds__(256) void linear_pq_kernel(
    const float* __restrict__ p, const float* __restrict__ q,
    const float* __restrict__ Ws, const float* __restrict__ bs,
    bf16_t* __restrict__ WP, bf16_t* __restrict__ WQ) {
    size_t tid = (size_t)blockIdx.x * 256 + threadIdx.x;     // KLAY*N*DM threads
    int e = (int)(tid & (DM - 1));
    size_t t = tid >> 7;
    int n = (int)(t & (NTOT - 1));
    int k = (int)(t >> 13);
    const float* w  = Ws + ((size_t)k * DM + e) * DM;        // Ws[k][e][:]
    const float* pr = p + (size_t)n * DM;
    const float* qr = q + (size_t)n * DM;
    float sp = 0.f, sq = 0.f;
    #pragma unroll 8
    for (int d = 0; d < DM; ++d) { float wv = w[d]; sp += pr[d] * wv; sq += qr[d] * wv; }
    float bias = bs[k * DM + e];
    size_t o = (size_t)n * (KLAY * DM) + (size_t)k * DM + e;
    WP[o] = f2bf(sp + bias);
    WQ[o] = f2bf(sq + bias);
}

// ------------------- dec = p_k[2] @ Wd^T + bd  -> bf16 ---------------------
__global__ __launch_bounds__(256) void dec_kernel(
    const float* __restrict__ p2, const float* __restrict__ Wd,
    const float* __restrict__ bd, bf16_t* __restrict__ dec) {
    size_t tid = (size_t)blockIdx.x * 256 + threadIdx.x;     // N*DM threads
    int e = (int)(tid & (DM - 1));
    int n = (int)(tid >> 7);
    const float* w  = Wd + (size_t)e * DM;
    const float* pr = p2 + (size_t)n * DM;
    float s = bd[e];
    #pragma unroll 8
    for (int d = 0; d < DM; ++d) s += pr[d] * w[d];
    dec[tid] = f2bf(s);
}

// ------------------------ WMMA fragment load from LDS ----------------------
// Tile is row-major [row][K] with stride LDS_STRIDE halves.
// bf16 16x32 A-layout: lanes 0-15 row M=lane, K in {0..7,16..23};
// lanes 16-31 row M=lane-16, K in {8..15,24..31}. B-operand uses the same
// per-lane packing with the tile stored K-minor per output column.
__device__ __forceinline__ v16bf load_frag(const bf16_t* tile, int row0, int lane) {
    const bf16_t* pp = tile + (size_t)(row0 + (lane & 15)) * LDS_STRIDE + ((lane >> 4) << 3);
    v8bf lo = *(const v8bf*)(pp);        // K = kh .. kh+7
    v8bf hi = *(const v8bf*)(pp + 16);   // K = 16+kh .. 16+kh+7
    v16bf f;
    #pragma unroll
    for (int i = 0; i < 8; ++i) { f[i] = lo[i]; f[i + 8] = hi[i]; }
    return f;
}

// ---------------------------------------------------------------------------
// Tiled bf16 WMMA GEMM, 128x128 block tile, BK=32, 256 threads (8 wave32).
// MODE 0: C = A^T(f32) @ WP(bf16 [K][384]); epi: /D_out, leaky -> p_k (f32)
// MODE 1: C = A  (f32) @ WQ(bf16 [K][384]); epi: /D_in,  leaky -> q_k + q2 bf16
// MODE 2: C = dec(bf16)@ q2^T(bf16);        epi: * a*d^b*exp(c*d) -> e_hat
// ---------------------------------------------------------------------------
template <int MODE>
__global__ __launch_bounds__(256) void gemm_wmma_k(
    const float* __restrict__ Aglob, const bf16_t* __restrict__ Abf,
    const bf16_t* __restrict__ Bbf, float* __restrict__ outF,
    const float* __restrict__ Dvec, bf16_t* __restrict__ q2out,
    const float* __restrict__ dist, const float* __restrict__ pa,
    const float* __restrict__ pb, const float* __restrict__ pc, int Kdim) {
    __shared__ bf16_t sA[128 * LDS_STRIDE];
    __shared__ bf16_t sB[128 * LDS_STRIDE];

    const int tid  = threadIdx.x;
    const int lane = tid & 31;
    const int wave = tid >> 5;
    const int wm   = wave >> 1;      // 0..3 -> 32-row strip
    const int wn   = wave & 1;       // 0..1 -> 64-col strip
    const int m0   = blockIdx.y * 128;
    const int n0   = blockIdx.x * 128;

    v8f acc[2][4];
    #pragma unroll
    for (int i = 0; i < 2; ++i)
        #pragma unroll
        for (int j = 0; j < 4; ++j)
            #pragma unroll
            for (int l = 0; l < 8; ++l) acc[i][j][l] = 0.f;

    for (int k0 = 0; k0 < Kdim; k0 += 32) {
        // ---- stage A tile -> sA[m][k] ----
        if constexpr (MODE == 0) {
            // transposed: sA[j][k] = A[k0+k][m0+j]  (f32 -> bf16, scatter stores)
            #pragma unroll
            for (int c = tid; c < 1024; c += 256) {
                int k  = c >> 5;
                int j4 = (c & 31) << 2;
                const float* gp = Aglob + (size_t)(k0 + k) * NTOT + (m0 + j4);
                v4f v = *(const v4f*)gp;
                if (k0 + 32 < Kdim)
                    __builtin_prefetch(gp + (size_t)32 * NTOT, 0, 1);  // next k-tile
                #pragma unroll
                for (int u = 0; u < 4; ++u) sA[(size_t)(j4 + u) * LDS_STRIDE + k] = f2bf(v[u]);
            }
        } else if constexpr (MODE == 1) {
            // direct: sA[m][k] = A[m0+m][k0+k]  (f32 -> packed bf16x4 store)
            #pragma unroll
            for (int c = tid; c < 1024; c += 256) {
                int m  = c >> 3;
                int k4 = (c & 7) << 2;
                const float* gp = Aglob + (size_t)(m0 + m) * NTOT + (k0 + k4);
                v4f v = *(const v4f*)gp;
                if (k0 + 32 < Kdim)
                    __builtin_prefetch(gp + 32, 0, 1);                 // next k-tile
                v4bf b = __builtin_convertvector(v, v4bf);
                *(v4bf*)(&sA[(size_t)m * LDS_STRIDE + k4]) = b;
            }
        } else {
            // direct bf16: sA[m][k] = dec[m0+m][k0+k]
            #pragma unroll
            for (int c = tid; c < 512; c += 256) {
                int m  = c >> 2;
                int k8 = (c & 3) << 3;
                v8bf v = *(const v8bf*)(Abf + (size_t)(m0 + m) * DM + (k0 + k8));
                *(v8bf*)(&sA[(size_t)m * LDS_STRIDE + k8]) = v;
            }
        }
        // ---- stage B tile (K-minor per column) -> sB[n][k] ----
        if constexpr (MODE == 0 || MODE == 1) {
            // sB[n][k] = B[k0+k][n0+n], source row stride 3*128
            #pragma unroll
            for (int c = tid; c < 512; c += 256) {
                int k  = c >> 4;
                int n8 = (c & 15) << 3;
                v8bf v = *(const v8bf*)(Bbf + (size_t)(k0 + k) * (KLAY * DM) + (n0 + n8));
                #pragma unroll
                for (int u = 0; u < 8; ++u) sB[(size_t)(n8 + u) * LDS_STRIDE + k] = v[u];
            }
        } else {
            // sB[n][k] = q2[n0+n][k0+k]  (already K-minor)
            #pragma unroll
            for (int c = tid; c < 512; c += 256) {
                int n  = c >> 2;
                int k8 = (c & 3) << 3;
                v8bf v = *(const v8bf*)(Bbf + (size_t)(n0 + n) * DM + (k0 + k8));
                *(v8bf*)(&sB[(size_t)n * LDS_STRIDE + k8]) = v;
            }
        }
        __syncthreads();

        v16bf af[2], bfv[4];
        #pragma unroll
        for (int mi = 0; mi < 2; ++mi) af[mi] = load_frag(sA, wm * 32 + mi * 16, lane);
        #pragma unroll
        for (int ni = 0; ni < 4; ++ni) bfv[ni] = load_frag(sB, wn * 64 + ni * 16, lane);
        #pragma unroll
        for (int mi = 0; mi < 2; ++mi)
            #pragma unroll
            for (int ni = 0; ni < 4; ++ni)
                acc[mi][ni] = __builtin_amdgcn_wmma_f32_16x16x32_bf16(
                    false, af[mi], false, bfv[ni], (short)0, acc[mi][ni], false, false);
        __syncthreads();
    }

    // ---- fused epilogue + store ----
    float sa = 0.f, sb = 0.f, sc = 0.f;
    if constexpr (MODE == 2) { sa = pa[0]; sb = pb[0]; sc = pc[0]; }
    #pragma unroll
    for (int mi = 0; mi < 2; ++mi)
        #pragma unroll
        for (int ni = 0; ni < 4; ++ni)
            #pragma unroll
            for (int v = 0; v < 8; ++v) {
                // C layout: VGPR v holds M=v (lanes 0-15) / M=v+8 (lanes 16-31)
                int row = m0 + wm * 32 + mi * 16 + v + ((lane >> 4) << 3);
                int col = n0 + wn * 64 + ni * 16 + (lane & 15);
                float val = acc[mi][ni][v];
                if constexpr (MODE == 0 || MODE == 1) {
                    val /= Dvec[row];
                    val = val > 0.f ? val : 0.01f * val;     // leaky_relu(0.01)
                    int kk = col >> 7, e = col & (DM - 1);
                    outF[(size_t)kk * (NTOT * DM) + (size_t)row * DM + e] = val;
                    if constexpr (MODE == 1) {
                        if (kk == 2) q2out[(size_t)row * DM + e] = f2bf(val);
                    }
                } else {
                    float d  = dist[(size_t)row * NTOT + col];
                    float fx = sa * __expf(sb * __logf(d) + sc * d);  // a*d^b*e^{c d}
                    outF[(size_t)row * NTOT + col] = val * fx;
                }
            }
}

// ---------------------------------------------------------------------------
extern "C" void kernel_launch(void* const* d_in, const int* in_sizes, int n_in,
                              void* d_out, int out_size, void* d_ws, size_t ws_size,
                              hipStream_t stream) {
    (void)in_sizes; (void)n_in; (void)out_size; (void)ws_size;
    const float* p    = (const float*)d_in[0];
    const float* q    = (const float*)d_in[1];
    const float* A    = (const float*)d_in[2];
    const float* dist = (const float*)d_in[3];
    const float* Ws   = (const float*)d_in[4];
    const float* bs   = (const float*)d_in[5];
    const float* Wd   = (const float*)d_in[6];
    const float* bd   = (const float*)d_in[7];
    const float* pa   = (const float*)d_in[8];
    const float* pb   = (const float*)d_in[9];
    const float* pc   = (const float*)d_in[10];

    float* out   = (float*)d_out;
    float* out_p = out;                                  // [3][N][128]
    float* out_q = out + (size_t)KLAY * NTOT * DM;       // [3][N][128]
    float* out_e = out + (size_t)2 * KLAY * NTOT * DM;   // [N][N]

    uint8_t* ws = (uint8_t*)d_ws;
    bf16_t* WP   = (bf16_t*)(ws);                        //  6,291,456 B
    bf16_t* WQ   = (bf16_t*)(ws + 6291456);              //  6,291,456 B
    bf16_t* decb = (bf16_t*)(ws + 12582912);             //  2,097,152 B
    bf16_t* q2b  = (bf16_t*)(ws + 14680064);             //  2,097,152 B
    float*  Dout = (float*)(ws + 16777216);              //     32,768 B
    float*  Din  = (float*)(ws + 16809984);              //     32,768 B

    colsum_kernel<<<NTOT / 256, 256, 0, stream>>>(A, Dout);
    rowsum_kernel<<<NTOT, 256, 0, stream>>>(A, Din);
    linear_pq_kernel<<<(KLAY * NTOT * DM) / 256, 256, 0, stream>>>(p, q, Ws, bs, WP, WQ);

    dim3 gBig(KLAY * DM / 128, NTOT / 128);              // (3, 64)
    gemm_wmma_k<0><<<gBig, 256, 0, stream>>>(A, nullptr, WP, out_p, Dout, nullptr,
                                             nullptr, nullptr, nullptr, nullptr, NTOT);
    gemm_wmma_k<1><<<gBig, 256, 0, stream>>>(A, nullptr, WQ, out_q, Din, q2b,
                                             nullptr, nullptr, nullptr, nullptr, NTOT);

    dec_kernel<<<(NTOT * DM) / 256, 256, 0, stream>>>(out_p + (size_t)2 * NTOT * DM,
                                                      Wd, bd, decb);

    dim3 gE(NTOT / 128, NTOT / 128);                     // (64, 64)
    gemm_wmma_k<2><<<gE, 256, 0, stream>>>(nullptr, decb, q2b, out_e, nullptr, nullptr,
                                           dist, pa, pb, pc, DM);
}
